// V12Model_56667798503548
// MI455X (gfx1250) — matrix operand order repaired and checked by reading, not compiled, not run on hardware
//
#include <hip/hip_runtime.h>
#include <math.h>

typedef __attribute__((ext_vector_type(16))) __bf16 v16bf;
typedef __attribute__((ext_vector_type(8)))  float  v8f;

// ---------------- helpers ----------------
static __device__ inline unsigned short f2bf(float f) {
    unsigned int u = __float_as_uint(f);
    unsigned int r = u + 0x7FFFu + ((u >> 16) & 1u);   // round-to-nearest-even bf16
    return (unsigned short)(r >> 16);
}
static __device__ inline float gelu_exact(float x) {
    return 0.5f * x * (1.0f + erff(x * 0.70710678118654752f));
}
static __device__ inline float wredsum(float v) {
    for (int o = 16; o > 0; o >>= 1) v += __shfl_xor(v, o, 32);
    return v;
}
static __device__ inline float wredmax(float v) {
    for (int o = 16; o > 0; o >>= 1) v = fmaxf(v, __shfl_xor(v, o, 32));
    return v;
}
static __device__ inline v16bf ldfrag(const unsigned short* p) {
    return *reinterpret_cast<const v16bf*>(p);
}
#define WMMA_BF16(a, b, c) \
    __builtin_amdgcn_wmma_f32_16x16x32_bf16(false, (a), false, (b), (short)0, (c), false, false)

// A-fragment element mapping (16x32 bf16): given (lane,slot) -> kk
//   kk = ((slot>>3)<<4) | ((lane>>4)<<3) | (slot&7),  m = lane&15
// B-fragment element mapping (32x16 bf16): given (lane,slot) -> kk
//   kk = ((slot>>3)<<4) | ((lane>>4)<<3) | (slot&7),                        n = lane&15

// ---------------- conv weight -> bf16 A-fragments ----------------
// out layout per layer: [Mtile(4)][kc(6)][lane(32)][slot(16)]
__global__ void prep_conv_w(const float* __restrict__ w, unsigned short* __restrict__ out) {
    int fi = blockIdx.x * 256 + threadIdx.x;       // < 12288
    int sl = fi & 15, ln = (fi >> 4) & 31;
    int kc = (fi >> 9) % 6;
    int Mt = fi / 3072;
    int m  = Mt * 16 + (ln & 15);
    int kk = ((sl >> 3) << 4) | ((ln >> 4) << 3) | (sl & 7);
    int K  = kc * 32 + kk;                         // K = k*64 + i, K<192 always
    int ci = K & 63, kt = K >> 6;
    out[fi] = f2bf(w[(m * 64 + ci) * 3 + kt]);
}

// ---------------- edge CNN megakernel ----------------
struct ConvParams { const float* g[5]; const float* b[5]; };

__global__ void __launch_bounds__(256)
edge_cnn(const float* __restrict__ seqs,    // [16384][8][64]
         const float* __restrict__ extra,   // [16384][3]
         const float* __restrict__ stem_w,  // [8][64]
         const float* __restrict__ stem_b,  // [64]
         const unsigned short* __restrict__ wfrag, // [5][4][6][512] bf16
         ConvParams cp,
         float* __restrict__ ep_in)         // [16384][67]
{
    __shared__ __align__(32) float xbuf[64][66];              // padded (col 0 and 65 zero)
    __shared__ __align__(32) unsigned short bsh[4][6][512];   // B fragments
    __shared__ __align__(32) float ybuf[64][64];
    const int seq  = blockIdx.x;
    const int t    = threadIdx.x;
    const int lane = t & 31, wave = t >> 5;

    // stem: x[d][l] = sum_c in[c][l] * stem_w[c][d] + stem_b[d]
    for (int j = 0; j < 16; ++j) {
        int idx = t * 16 + j;                 // 4096
        int d = idx >> 6, l = idx & 63;
        float s = stem_b[d];
        for (int c = 0; c < 8; ++c)
            s += seqs[seq * 512 + c * 64 + l] * stem_w[c * 64 + d];
        xbuf[d][l + 1] = s;
    }
    if (t < 64) { xbuf[t][0] = 0.f; xbuf[t][65] = 0.f; }

    for (int layer = 0; layer < 5; ++layer) {
        __syncthreads();
        // build B fragments: X[K=(k*64+i)][l] = xpad[i][l+k]
        for (int j = 0; j < 48; ++j) {
            int fi = t * 48 + j;              // < 12288
            int sl = fi & 15, ln = (fi >> 4) & 31;
            int kc = (fi >> 9) % 6;
            int nt = fi / 3072;
            int n  = nt * 16 + (ln & 15);     // l
            int kk = ((sl >> 3) << 4) | ((ln >> 4) << 3) | (sl & 7);
            int K  = kc * 32 + kk;
            int ci = K & 63, kt = K >> 6;
            float v = (K < 192) ? xbuf[ci][n + kt] : 0.f;
            bsh[nt][kc][(ln << 4) | sl] = f2bf(v);
        }
        __syncthreads();
        // 16 output tiles of 16x16, 8 waves -> 2 tiles each, K = 6 chunks of 32
        for (int tt = 0; tt < 2; ++tt) {
            int tile = wave * 2 + tt;
            int Mt = tile >> 2, Nt = tile & 3;
            v8f acc = {};
            for (int kc = 0; kc < 6; ++kc) {
                v16bf a = ldfrag(&wfrag[(((layer * 4 + Mt) * 6 + kc) << 9) + lane * 16]);
                v16bf b = ldfrag(&bsh[Nt][kc][lane * 16]);
                acc = WMMA_BF16(a, b, acc);
            }
            for (int r = 0; r < 8; ++r) {
                int m = Mt * 16 + r + ((lane & 16) ? 8 : 0);
                int n = Nt * 16 + (lane & 15);
                ybuf[m][n] = acc[r];
            }
        }
        __syncthreads();
        // GroupNorm(8 groups) + GELU + residual; wave g handles group g
        {
            int g = wave;
            float s = 0.f, s2 = 0.f;
            for (int j = 0; j < 16; ++j) {
                int e = lane * 16 + j;                // 0..511
                int ch = g * 8 + (e >> 6), l = e & 63;
                float v = ybuf[ch][l];
                s += v; s2 += v * v;
            }
            s = wredsum(s); s2 = wredsum(s2);
            float mean = s * (1.f / 512.f);
            float var  = s2 * (1.f / 512.f) - mean * mean;
            float inv  = rsqrtf(var + 1e-5f);
            for (int j = 0; j < 16; ++j) {
                int e = lane * 16 + j;
                int ch = g * 8 + (e >> 6), l = e & 63;
                float v = (ybuf[ch][l] - mean) * inv * cp.g[layer][ch] + cp.b[layer][ch];
                xbuf[ch][l + 1] += gelu_exact(v);
            }
        }
    }
    __syncthreads();
    // mean over L, write concat row [cnn(64) | extra(3)]
    if (t < 64) {
        float s = 0.f;
        for (int l = 0; l < 64; ++l) s += xbuf[t][l + 1];
        ep_in[seq * 67 + t] = s * (1.f / 64.f);
    } else if (t < 67) {
        ep_in[seq * 67 + t] = extra[seq * 3 + (t - 64)];
    }
}

// ---------------- generic bf16-WMMA GEMM ----------------
// C[M,N] = maybe_gelu(A[M,lda] @ W[K,N] + bias) + res ;  fp32 in/out
__global__ void __launch_bounds__(128)
gemm_bf16(const float* __restrict__ A, int lda,
          const float* __restrict__ W,
          const float* __restrict__ bias,
          const float* __restrict__ res,
          float* __restrict__ C,
          int M, int N, int K, int act_gelu)
{
    __shared__ __align__(32) unsigned short Ash[512];
    __shared__ __align__(32) unsigned short Bsh[4][512];
    const int t = threadIdx.x;
    const int lane = t & 31, wave = t >> 5;
    const int m0 = blockIdx.x * 16, n0 = blockIdx.y * 64;
    v8f acc = {};
    const int kch = (K + 31) >> 5;
    for (int kc = 0; kc < kch; ++kc) {
        __syncthreads();
        for (int j = 0; j < 4; ++j) {                // stage A (512 halfs)
            int fi = t * 4 + j;
            int ln = fi >> 4, sl = fi & 15;
            int m  = m0 + (ln & 15);
            int kk = ((sl >> 3) << 4) | ((ln >> 4) << 3) | (sl & 7);
            int k  = kc * 32 + kk;
            float v = (m < M && k < K) ? A[m * lda + k] : 0.f;
            Ash[fi] = f2bf(v);
        }
        for (int j = 0; j < 16; ++j) {               // stage B (2048 halfs)
            int fi = t * 16 + j;
            int nt = fi >> 9, r = fi & 511;
            int ln = r >> 4, sl = r & 15;
            int n  = n0 + nt * 16 + (ln & 15);
            int kk = ((sl >> 3) << 4) | ((ln >> 4) << 3) | (sl & 7);
            int k  = kc * 32 + kk;
            float v = (n < N && k < K) ? W[k * N + n] : 0.f;
            Bsh[nt][r] = f2bf(v);
        }
        __syncthreads();
        v16bf a = ldfrag(&Ash[lane * 16]);
        v16bf b = ldfrag(&Bsh[wave][lane * 16]);
        acc = WMMA_BF16(a, b, acc);
    }
    for (int r = 0; r < 8; ++r) {
        int m = m0 + r + ((lane & 16) ? 8 : 0);
        int n = n0 + wave * 16 + (lane & 15);
        if (m < M && n < N) {
            float v = acc[r];
            if (bias) v += bias[n];
            if (act_gelu) v = gelu_exact(v);
            if (res) v += res[m * N + n];
            C[m * N + n] = v;
        }
    }
}

// ---------------- LayerNorm (optionally + GELU); wave per row ----------------
__global__ void ln_kernel(const float* __restrict__ x, const float* __restrict__ g,
                          const float* __restrict__ b, float* __restrict__ out,
                          int rows, int dim, int do_gelu)
{
    int lane = threadIdx.x & 31, wave = threadIdx.x >> 5;
    int row = blockIdx.x * 8 + wave;
    if (row >= rows) return;
    const float* xr = x + (size_t)row * dim;
    float s = 0.f, s2 = 0.f;
    for (int i = lane; i < dim; i += 32) { float v = xr[i]; s += v; s2 += v * v; }
    s = wredsum(s); s2 = wredsum(s2);
    float mean = s / (float)dim;
    float var  = s2 / (float)dim - mean * mean;
    float inv  = rsqrtf(var + 1e-5f);
    float* orow = out + (size_t)row * dim;
    for (int i = lane; i < dim; i += 32) {
        float v = (xr[i] - mean) * inv * g[i] + b[i];
        if (do_gelu) v = gelu_exact(v);
        orow[i] = v;
    }
}

// ---------------- edge bias: ebias[(b*8+h)*4096 + i*64 + j] ----------------
__global__ void ebias_kernel(const float* __restrict__ e, const float* __restrict__ ew,
                             const float* __restrict__ eb, float* __restrict__ out)
{
    int tid = blockIdx.x * 256 + threadIdx.x;     // < 131072
    int j = tid & 63, i = (tid >> 6) & 63, h = (tid >> 12) & 7, b = tid >> 15;
    const float* er = e + (size_t)(((b * 64 + i) * 64 + j)) * 64;
    float s = eb[h];
    for (int k = 0; k < 64; ++k) s += er[k] * ew[k * 8 + h];
    out[tid] = s;
}

// ---------------- h += type_emb[node_types] ----------------
__global__ void add_emb(float* __restrict__ h, const float* __restrict__ emb,
                        const int* __restrict__ types)
{
    int tid = blockIdx.x * 256 + threadIdx.x;     // 65536
    int tk = tid >> 8, d = tid & 255;
    h[tid] += emb[types[tk] * 256 + d];
}

// ---------------- attention per (b,h): WMMA scores + softmax + WMMA AV ----------------
__global__ void __launch_bounds__(256)
attn_kernel(const float* __restrict__ q, const float* __restrict__ k,
            const float* __restrict__ v, const float* __restrict__ ebias,
            const unsigned char* __restrict__ mask, float* __restrict__ o)
{
    __shared__ __align__(32) unsigned short afr[4][2][512];
    __shared__ __align__(32) unsigned short bfr[4][512];
    __shared__ float sbuf[64][64];
    const int bh = blockIdx.x, b = bh >> 3, h = bh & 7;
    const int t = threadIdx.x, lane = t & 31, wave = t >> 5;

    // stage q -> A frags [4][0], k^T -> B frags [4]
    for (int j = 0; j < 8; ++j) {
        int fi = t * 8 + j;                       // 2048
        int Mt = fi >> 9, r = fi & 511, ln = r >> 4, sl = r & 15;
        int m  = Mt * 16 + (ln & 15);
        int kkA = ((sl >> 3) << 4) | ((ln >> 4) << 3) | (sl & 7);  // d 0..31
        afr[Mt][0][r] = f2bf(q[((b * 64 + m) << 8) + h * 32 + kkA]);
        int kkB = ((sl >> 3) << 4) | ((ln >> 4) << 3) | (sl & 7);  // d 0..31
        bfr[Mt][r] = f2bf(k[((b * 64 + m) << 8) + h * 32 + kkB]);  // B[d][j]=k[j][d]
    }
    __syncthreads();
    const float scale = 0.17677669529663689f;     // 1/sqrt(32)
    for (int tt = 0; tt < 2; ++tt) {
        int tile = wave * 2 + tt;
        int Mt = tile >> 2, Nt = tile & 3;
        v8f acc = {};
        v16bf a  = ldfrag(&afr[Mt][0][lane * 16]);
        v16bf bm = ldfrag(&bfr[Nt][lane * 16]);
        acc = WMMA_BF16(a, bm, acc);
        for (int r = 0; r < 8; ++r) {
            int i = Mt * 16 + r + ((lane & 16) ? 8 : 0);
            int j = Nt * 16 + (lane & 15);
            sbuf[i][j] = acc[r] * scale + ebias[(bh << 12) + (i << 6) + j];
        }
    }
    __syncthreads();
    // masked softmax rows (nan_to_num -> zeros when fully masked)
    for (int rr = 0; rr < 8; ++rr) {
        int i = wave * 8 + rr;
        float v0 = sbuf[i][lane], v1 = sbuf[i][lane + 32];
        if (!mask[b * 64 + lane])      v0 = -INFINITY;
        if (!mask[b * 64 + lane + 32]) v1 = -INFINITY;
        float mx = wredmax(fmaxf(v0, v1));
        float e0 = 0.f, e1 = 0.f;
        if (mx != -INFINITY) { e0 = expf(v0 - mx); e1 = expf(v1 - mx); }
        float sum = wredsum(e0 + e1);
        float inv = (sum > 0.f) ? 1.f / sum : 0.f;
        sbuf[i][lane] = e0 * inv; sbuf[i][lane + 32] = e1 * inv;
    }
    __syncthreads();
    // stage attn -> A frags [4][2], v -> B frags [2][2]
    for (int j = 0; j < 16; ++j) {
        int fi = t * 16 + j;                      // 4096
        int Mt = fi >> 10, kc = (fi >> 9) & 1, r = fi & 511;
        int ln = r >> 4, sl = r & 15;
        int m  = Mt * 16 + (ln & 15);
        int kk = ((sl >> 3) << 4) | ((ln >> 4) << 3) | (sl & 7);
        afr[Mt][kc][r] = f2bf(sbuf[m][kc * 32 + kk]);
    }
    unsigned short* vfr = &bfr[0][0];             // reuse as [2][2][512]
    for (int j = 0; j < 8; ++j) {
        int fi = t * 8 + j;                       // 2048
        int Nt = fi >> 10, kc = (fi >> 9) & 1, r = fi & 511;
        int ln = r >> 4, sl = r & 15;
        int n  = Nt * 16 + (ln & 15);             // d 0..31
        int kk = ((sl >> 3) << 4) | ((ln >> 4) << 3) | (sl & 7);
        int K  = kc * 32 + kk;                    // j 0..63
        vfr[(((Nt << 1) | kc) << 9) + r] = f2bf(v[((b * 64 + K) << 8) + h * 32 + n]);
    }
    __syncthreads();
    {
        int Mt = wave >> 1, Nt = wave & 1;        // 8 tiles, 1 per wave
        v8f acc = {};
        for (int kc = 0; kc < 2; ++kc) {
            v16bf a  = ldfrag(&afr[Mt][kc][lane * 16]);
            v16bf bm = ldfrag(&vfr[(((Nt << 1) | kc) << 9) + lane * 16]);
            acc = WMMA_BF16(a, bm, acc);
        }
        for (int r = 0; r < 8; ++r) {
            int m = Mt * 16 + r + ((lane & 16) ? 8 : 0);
            int n = Nt * 16 + (lane & 15);
            o[((b * 64 + m) << 8) + h * 32 + n] = acc[r];
        }
    }
}

// ---------------- host ----------------
extern "C" void kernel_launch(void* const* d_in, const int* in_sizes, int n_in,
                              void* d_out, int out_size, void* d_ws, size_t ws_size,
                              hipStream_t stream) {
    (void)in_sizes; (void)n_in; (void)out_size; (void)ws_size;
    auto F = [&](int i) { return (const float*)d_in[i]; };

    const float* edge_seqs  = F(0);
    const float* extra      = F(1);
    const float* node_feats = F(2);
    const int*   node_types = (const int*)d_in[3];
    const unsigned char* mask = (const unsigned char*)d_in[4];
    const float* stem_w = F(5);
    const float* stem_b = F(6);
    const float *conv_w[5]; ConvParams cp;
    for (int i = 0; i < 5; ++i) { conv_w[i] = F(7 + 3 * i); cp.g[i] = F(8 + 3 * i); cp.b[i] = F(9 + 3 * i); }
    const float* ep_w    = F(22); const float* ep_b    = F(23);
    const float* ep_ln_g = F(24); const float* ep_ln_b = F(25);
    const float* nm_w1   = F(26); const float* nm_b1   = F(27);
    const float* nm_ln_g = F(28); const float* nm_ln_b = F(29);
    const float* nm_w2   = F(30); const float* nm_b2   = F(31);
    const float* type_emb= F(32);
    const int LB = 33;            // layer base, 15 entries per layer
    const float* fin_g = F(123); const float* fin_b = F(124);
    const float* hd_w1 = F(125); const float* hd_b1 = F(126);
    const float* hd_w2 = F(127); const float* hd_b2 = F(128);

    // workspace carve-up (256B aligned)
    char* ws = (char*)d_ws;
    size_t off = 0;
    auto alloc = [&](size_t bytes) { char* p = ws + off; off = (off + bytes + 255) & ~(size_t)255; return p; };
    unsigned short* wfrag = (unsigned short*)alloc(5 * 12288 * 2);
    float* ep_in = (float*)alloc((size_t)16384 * 67 * 4);
    float* etmp  = (float*)alloc((size_t)16384 * 64 * 4);
    float* ebuf  = (float*)alloc((size_t)16384 * 64 * 4);
    float* ebias = (float*)alloc((size_t)131072 * 4);
    float* hbuf  = (float*)alloc(256 * 256 * 4);
    float* xnbuf = (float*)alloc(256 * 256 * 4);
    float* qb    = (float*)alloc(256 * 256 * 4);
    float* kb    = (float*)alloc(256 * 256 * 4);
    float* vb    = (float*)alloc(256 * 256 * 4);
    float* ob    = (float*)alloc(256 * 256 * 4);
    float* ffb   = (float*)alloc(256 * 1024 * 4);
    float* ntmp  = (float*)alloc(256 * 256 * 4);
    float* ntmp2 = (float*)alloc(256 * 256 * 4);
    float* t2    = (float*)alloc(256 * 128 * 4);

    // 1) conv weights -> bf16 fragments
    for (int l = 0; l < 5; ++l)
        prep_conv_w<<<48, 256, 0, stream>>>(conv_w[l], wfrag + l * 12288);
    // 2) edge CNN (stem + 5 conv blocks + pool + concat) : 16384 blocks
    edge_cnn<<<16384, 256, 0, stream>>>(edge_seqs, extra, stem_w, stem_b, wfrag, cp, ep_in);
    // 3) edge projector + LN + GELU
    gemm_bf16<<<dim3(1024, 1), 128, 0, stream>>>(ep_in, 67, ep_w, ep_b, nullptr, etmp, 16384, 64, 67, 0);
    ln_kernel<<<2048, 256, 0, stream>>>(etmp, ep_ln_g, ep_ln_b, ebuf, 16384, 64, 1);
    // 4) node encoder
    gemm_bf16<<<dim3(16, 4), 128, 0, stream>>>(node_feats, 7, nm_w1, nm_b1, nullptr, ntmp, 256, 256, 7, 0);
    ln_kernel<<<32, 256, 0, stream>>>(ntmp, nm_ln_g, nm_ln_b, ntmp2, 256, 256, 1);
    gemm_bf16<<<dim3(16, 4), 128, 0, stream>>>(ntmp2, 256, nm_w2, nm_b2, nullptr, hbuf, 256, 256, 256, 0);
    add_emb<<<256, 256, 0, stream>>>(hbuf, type_emb, node_types);
    // 5) transformer layers
    for (int l = 0; l < 6; ++l) {
        int base = LB + 15 * l;
        const float* n1_g = F(base + 0); const float* n1_b = F(base + 1);
        const float* wq   = F(base + 2); const float* wk   = F(base + 3);
        const float* wv   = F(base + 4); const float* wo   = F(base + 5);
        const float* wo_b = F(base + 6);
        const float* eb_w = F(base + 7); const float* eb_b = F(base + 8);
        const float* n2_g = F(base + 9); const float* n2_b = F(base + 10);
        const float* fw1  = F(base + 11); const float* fb1 = F(base + 12);
        const float* fw2  = F(base + 13); const float* fb2 = F(base + 14);

        ln_kernel<<<32, 256, 0, stream>>>(hbuf, n1_g, n1_b, xnbuf, 256, 256, 0);
        gemm_bf16<<<dim3(16, 4), 128, 0, stream>>>(xnbuf, 256, wq, nullptr, nullptr, qb, 256, 256, 256, 0);
        gemm_bf16<<<dim3(16, 4), 128, 0, stream>>>(xnbuf, 256, wk, nullptr, nullptr, kb, 256, 256, 256, 0);
        gemm_bf16<<<dim3(16, 4), 128, 0, stream>>>(xnbuf, 256, wv, nullptr, nullptr, vb, 256, 256, 256, 0);
        ebias_kernel<<<512, 256, 0, stream>>>(ebuf, eb_w, eb_b, ebias);
        attn_kernel<<<32, 256, 0, stream>>>(qb, kb, vb, ebias, mask, ob);
        gemm_bf16<<<dim3(16, 4), 128, 0, stream>>>(ob, 256, wo, wo_b, hbuf, hbuf, 256, 256, 256, 0);
        ln_kernel<<<32, 256, 0, stream>>>(hbuf, n2_g, n2_b, xnbuf, 256, 256, 0);
        gemm_bf16<<<dim3(16, 16), 128, 0, stream>>>(xnbuf, 256, fw1, fb1, nullptr, ffb, 256, 1024, 256, 1);
        gemm_bf16<<<dim3(16, 4), 128, 0, stream>>>(ffb, 1024, fw2, fb2, hbuf, hbuf, 256, 256, 1024, 0);
    }
    // 6) final norm + head
    ln_kernel<<<32, 256, 0, stream>>>(hbuf, fin_g, fin_b, xnbuf, 256, 256, 0);
    gemm_bf16<<<dim3(16, 2), 128, 0, stream>>>(xnbuf, 256, hd_w1, hd_b1, nullptr, t2, 256, 128, 256, 1);
    gemm_bf16<<<dim3(16, 1), 128, 0, stream>>>(t2, 128, hd_w2, hd_b2, nullptr, (float*)d_out, 256, 8, 128, 0);
}
